// ActionModule_71390946394746
// MI455X (gfx1250) — matrix operand order repaired and checked by reading, not compiled, not run on hardware
//
#include <hip/hip_runtime.h>
#include <hip/hip_bf16.h>

// ---------------- constants from the reference ----------------
#define S_DIM   1560      // th*tw
#define T_DIM   16        // tt
#define L_DIM   24960     // S*T
#define IMG_C   1536
#define CM_C    1024
#define QKV_C   3072
#define HD_N    64
#define LOCAL_W 6

typedef __attribute__((ext_vector_type(16))) __bf16        v16bf;
typedef __attribute__((ext_vector_type(8)))  float         v8f;

// ---------------- bf16 helpers (raw ushort storage) ----------------
__device__ inline unsigned short f2bf(float f) {
  unsigned int u = __builtin_bit_cast(unsigned int, f);
  u += 0x7FFFu + ((u >> 16) & 1u);           // round-to-nearest-even
  return (unsigned short)(u >> 16);
}
__device__ inline float bf2f(unsigned short h) {
  unsigned int u = ((unsigned int)h) << 16;
  return __builtin_bit_cast(float, u);
}

// branchless, overflow-safe tanh: 1 - 2/(e^{2x}+1)  (e->inf => 1, e->0 => -1)
__device__ inline float tanh_fast(float x) {
  float e = __expf(2.0f * x);
  return 1.0f - 2.0f / (e + 1.0f);
}
__device__ inline float gelu_tanh(float x) {
  float x3 = x * x * x;
  return 0.5f * x * (1.0f + tanh_fast(0.7978845608028654f * (x + 0.044715f * x3)));
}

// rope: only first 4 pairs rotate; freqs = 4^-p  (THETA=256)
__device__ inline void rope_apply(float* x, float t) {
  #pragma unroll
  for (int p = 0; p < 4; ++p) {
    float freq = exp2f(-2.0f * (float)p);
    float ang = t * freq;
    float c = cosf(ang), sn = sinf(ang);
    float x0 = x[2*p], x1 = x[2*p+1];
    x[2*p]   = x0 * c - x1 * sn;
    x[2*p+1] = x1 * c + x0 * sn;
  }
}

// ---------------- small prep kernels ----------------
// transposed bf16 weight: dst[n*K + k] = bf16(src[k*N + n])   (output-coalesced)
__global__ __launch_bounds__(256) void cvt_transpose_kernel(const float* __restrict__ src,
                                                            unsigned short* __restrict__ dst,
                                                            int K, int N) {
  int i = blockIdx.x * 256 + threadIdx.x;
  if (i >= K * N) return;
  int n = i / K, k = i - n * K;
  dst[i] = f2bf(src[(size_t)k * N + n]);
}

// xb[(s*16+t)*1536 + c] = bf16( x[(t*1560+s)*1536 + c] )
__global__ __launch_bounds__(256) void permute_x_kernel(const float* __restrict__ x,
                                                        unsigned short* __restrict__ xb) {
  long long i = (long long)blockIdx.x * 256 + threadIdx.x;
  if (i >= (long long)L_DIM * IMG_C) return;
  int c = (int)(i % IMG_C);
  int r = (int)(i / IMG_C);
  int s = r >> 4, t = r & 15;
  xb[i] = f2bf(x[((size_t)t * S_DIM + s) * IMG_C + c]);
}

// tbias[t][n] = mm_b1[n] + sum_{a<12,c<2} mouse[clamp(4t+a-12)][c] * mm_w1[1536+a*2+c][n]
__global__ __launch_bounds__(256) void tbias_kernel(const float* __restrict__ mouse,
                                                    const float* __restrict__ w1,
                                                    const float* __restrict__ b1,
                                                    float* __restrict__ tbias) {
  int gid = blockIdx.x * 256 + threadIdx.x;   // 16*1024
  int t = gid >> 10, n = gid & 1023;
  float acc = b1[n];
  #pragma unroll
  for (int a = 0; a < 12; ++a) {
    int f = 4 * t + a - 12; if (f < 0) f = 0;
    acc += mouse[f*2 + 0] * w1[(size_t)(1536 + a*2 + 0) * CM_C + n];
    acc += mouse[f*2 + 1] * w1[(size_t)(1536 + a*2 + 1) * CM_C + n];
  }
  tbias[gid] = acc;
}

// keyboard MLP stage 1: kbh[f][h] = silu(kc[f]@kb_w1 + b1)
__global__ __launch_bounds__(256) void kb1_kernel(const float* __restrict__ kc,
                                                  const float* __restrict__ w1, const float* __restrict__ b1,
                                                  float* __restrict__ kbh) {
  int gid = blockIdx.x * 256 + threadIdx.x;
  if (gid >= 61 * 128) return;
  int f = gid >> 7, h = gid & 127;
  float acc = b1[h];
  #pragma unroll
  for (int j = 0; j < 6; ++j) acc += kc[f*6 + j] * w1[j*128 + h];
  kbh[gid] = acc / (1.0f + __expf(-acc));      // silu
}
// stage 2: kbv = kbh @ kb_w2 + b2
__global__ __launch_bounds__(256) void kb2_kernel(const float* __restrict__ kbh,
                                                  const float* __restrict__ w2, const float* __restrict__ b2,
                                                  float* __restrict__ kbv) {
  int gid = blockIdx.x * 256 + threadIdx.x;
  if (gid >= 61 * 128) return;
  int f = gid >> 7, h = gid & 127;
  float acc = b2[h];
  for (int j = 0; j < 128; ++j) acc += kbh[f*128 + j] * w2[j*128 + h];
  kbv[gid] = acc;
}

// kv = gk @ key_kv_w : out (16, 2048) split into k2raw/v2 each [t][h][d]
__global__ __launch_bounds__(256) void kv_kernel(const float* __restrict__ kbv,
                                                 const float* __restrict__ wkv,
                                                 float* __restrict__ k2raw, float* __restrict__ v2) {
  int gid = blockIdx.x * 256 + threadIdx.x;   // 16*2048
  int t = gid >> 11, n = gid & 2047;
  float acc = 0.f;
  for (int a = 0; a < 12; ++a) {
    int f = 4 * t + a - 12; if (f < 0) f = 0;
    const float* kb = kbv + f * 128;
    const float* w  = wkv + (size_t)(a * 128) * 2048 + n;
    for (int cc = 0; cc < 128; ++cc) acc += kb[cc] * w[(size_t)cc * 2048];
  }
  int kvi = n >> 10, h = (n >> 6) & 15, d = n & 63;
  float* dst = kvi ? v2 : k2raw;
  dst[(t * 16 + h) * 64 + d] = acc;
}

// rms + rope on k2 (per (t,h) row of 64)
__global__ __launch_bounds__(256) void k2rope_kernel(const float* __restrict__ k2raw,
                                                     const float* __restrict__ kkw,
                                                     float* __restrict__ k2r) {
  int tid = threadIdx.x;                      // single block of 256
  int t = tid >> 4, h = tid & 15;
  float x[64]; float ss = 0.f;
  #pragma unroll
  for (int d = 0; d < 64; ++d) { x[d] = k2raw[(t*16+h)*64 + d]; ss += x[d]*x[d]; }
  float inv = rsqrtf(ss * (1.0f/64.0f) + 1e-6f);
  #pragma unroll
  for (int d = 0; d < 64; ++d) x[d] *= inv * kkw[d];
  rope_apply(x, (float)t);
  #pragma unroll
  for (int d = 0; d < 64; ++d) k2r[(t*16+h)*64 + d] = x[d];
}

// ---------------- WMMA GEMM: C[M,N] = epilogue(A[M,K](bf16) @ Wt[N,K](bf16)^T) ----------------
// block 128x128x32, 8 waves, wave tile 32x64 (2x4 wmma 16x16 tiles)
// Wt is pre-transposed, so A and B tile loaders are identical (pure b128 traffic).
// double-buffered LDS, one barrier per K-step, global prefetch of K+64 tile.
__device__ inline v16bf frag_ld(const unsigned short* p0, const unsigned short* p1) {
  union U { struct { uint4 lo, hi; } q; v16bf v; } u;
  u.q.lo = *(const uint4*)p0;
  u.q.hi = *(const uint4*)p1;
  return u.v;
}

template<int ACT, int MODE, bool HAS_BIAS, bool HAS_ROWBIAS>
__global__ __launch_bounds__(256) void gemm_bf16_wmma(
    const unsigned short* __restrict__ A, const unsigned short* __restrict__ Wt,
    const float* __restrict__ bias, const float* __restrict__ rowbias,
    unsigned short* __restrict__ outb, float* __restrict__ outf,
    const float* __restrict__ resid,
    int M, int N, int K)
    // MODE 0: bf16 out at [r,c]
    // MODE 1: remap r=(s*16+t)->rr=t*S+s, v+=resid[rr,c], outf[rr,c]=v, outb[rr,c]=bf16(v)
    // MODE 2: v+=resid[r,c], outf[r,c]=v
{
  __shared__ unsigned short As[2][128 * 32];  // [row][k] row-major
  __shared__ unsigned short Bs[2][128 * 32];  // [col][k] row-major (Wt layout)
  const int tid = threadIdx.x;
  const int lane = tid & 31, wave = tid >> 5;
  const int waveM = wave & 3, waveN = wave >> 2;
  const int row0 = blockIdx.y * 128, col0 = blockIdx.x * 128;
  const int l16 = lane & 15, lh = lane >> 4;

  // per-thread tile-copy role (identical for A and B): 16 bf16 = 2 x b128
  const int cr = tid >> 1, chf = tid & 1;

  auto loadT = [&](const unsigned short* base, int r0, int k0, uint4& t0, uint4& t1) {
    const uint4* src = (const uint4*)(base + (size_t)(r0 + cr) * K + k0 + chf * 16);
    t0 = src[0]; t1 = src[1];
  };
  auto storeT = [&](unsigned short* lds, uint4 t0, uint4 t1) {
    *(uint4*)&lds[cr * 32 + chf * 16]     = t0;
    *(uint4*)&lds[cr * 32 + chf * 16 + 8] = t1;
  };

  v8f acc[2][4] = {};

  uint4 a0, a1, b0, b1;
  loadT(A,  row0, 0, a0, a1);
  loadT(Wt, col0, 0, b0, b1);
  storeT(As[0], a0, a1);
  storeT(Bs[0], b0, b1);
  __syncthreads();

  int buf = 0;
  for (int k0 = 0; k0 < K; k0 += 32) {
    const int kn = k0 + 32;
    const bool hasNext = kn < K;
    if (hasNext) {                         // overlap next tile's HBM loads with WMMA
      loadT(A,  row0, kn, a0, a1);
      loadT(Wt, col0, kn, b0, b1);
      if (kn + 32 < K) {                   // gfx1250 global_prefetch_b8 for K+64 tile
        __builtin_prefetch(A  + (size_t)(row0 + cr) * K + kn + 32 + chf * 16, 0, 1);
        __builtin_prefetch(Wt + (size_t)(col0 + cr) * K + kn + 32 + chf * 16, 0, 1);
      }
    }

    v16bf af[2], bfr[4];
    #pragma unroll
    for (int mt = 0; mt < 2; ++mt) {
      int mrow = waveM * 32 + mt * 16 + l16;
      af[mt] = frag_ld(&As[buf][mrow * 32 + lh * 16], &As[buf][mrow * 32 + lh * 16 + 8]);
    }
    #pragma unroll
    for (int nt = 0; nt < 4; ++nt) {
      int col = waveN * 64 + nt * 16 + l16;
      bfr[nt] = frag_ld(&Bs[buf][col * 32 + lh * 16], &Bs[buf][col * 32 + lh * 16 + 8]);
    }
    #pragma unroll
    for (int mt = 0; mt < 2; ++mt)
      #pragma unroll
      for (int nt = 0; nt < 4; ++nt)
        acc[mt][nt] = __builtin_amdgcn_wmma_f32_16x16x32_bf16(
            false, af[mt], false, bfr[nt], (short)0, acc[mt][nt], false, false);

    if (hasNext) {
      storeT(As[buf ^ 1], a0, a1);
      storeT(Bs[buf ^ 1], b0, b1);
      __syncthreads();                     // single barrier per K-step
    }
    buf ^= 1;
  }

  // epilogue: lane l16 = N column, VGPR i -> local row i + 8*lh
  #pragma unroll
  for (int mt = 0; mt < 2; ++mt) {
    #pragma unroll
    for (int nt = 0; nt < 4; ++nt) {
      v8f a = acc[mt][nt];
      int c = col0 + waveN * 64 + nt * 16 + l16;
      float bcol = HAS_BIAS ? bias[c] : 0.0f;
      #pragma unroll
      for (int i = 0; i < 8; ++i) {
        int r = row0 + waveM * 32 + mt * 16 + lh * 8 + i;
        float v = a[i] + bcol;
        if (HAS_ROWBIAS) v += rowbias[(r & 15) * N + c];
        if (ACT == 1) v = gelu_tanh(v);
        if (MODE == 0) {
          outb[(size_t)r * N + c] = f2bf(v);
        } else if (MODE == 1) {
          size_t rr = (size_t)(r & 15) * S_DIM + (r >> 4);
          size_t idx = rr * N + c;
          v += resid[idx];
          outf[idx] = v;
          outb[idx] = f2bf(v);
        } else {
          size_t idx = (size_t)r * N + c;
          v += resid[idx];
          outf[idx] = v;
        }
      }
    }
  }
}

// ---------------- LayerNorm over 1024, one block per row ----------------
__global__ __launch_bounds__(256) void ln_kernel(const unsigned short* __restrict__ h2,
                                                 const float* __restrict__ g, const float* __restrict__ b,
                                                 unsigned short* __restrict__ out) {
  int row = blockIdx.x, tid = threadIdx.x;
  const unsigned short* p = h2 + (size_t)row * CM_C;
  float v[4]; float s = 0.f, s2 = 0.f;
  #pragma unroll
  for (int j = 0; j < 4; ++j) { v[j] = bf2f(p[tid * 4 + j]); s += v[j]; s2 += v[j] * v[j]; }
  __shared__ float rs[256], rq[256];
  rs[tid] = s; rq[tid] = s2;
  __syncthreads();
  for (int off = 128; off > 0; off >>= 1) {
    if (tid < off) { rs[tid] += rs[tid + off]; rq[tid] += rq[tid + off]; }
    __syncthreads();
  }
  float mean = rs[0] * (1.0f / CM_C);
  float var  = rq[0] * (1.0f / CM_C) - mean * mean;
  float inv  = rsqrtf(var + 1e-5f);
  #pragma unroll
  for (int j = 0; j < 4; ++j) {
    int c = tid * 4 + j;
    out[(size_t)row * CM_C + c] = f2bf((v[j] - mean) * inv * g[c] + b[c]);
  }
}

// ---------------- attention 1: per s, thread = (head, t), T=16, local window ----------------
__global__ __launch_bounds__(256) void attn1_kernel(const unsigned short* __restrict__ qkv,
                                                    const float* __restrict__ mqw, const float* __restrict__ mkw,
                                                    unsigned short* __restrict__ out) {
  __shared__ unsigned short kk[16 * 16 * 64];   // roped k, bf16
  __shared__ unsigned short vv[16 * 16 * 64];   // v, bf16
  int s = blockIdx.x, tid = threadIdx.x;
  int h = tid >> 4, tk = tid & 15;
  size_t rowb = (size_t)(s * 16 + tk) * QKV_C;
  {
    float x[64]; float ss = 0.f;
    #pragma unroll
    for (int d = 0; d < 64; ++d) { x[d] = bf2f(qkv[rowb + 1024 + h * 64 + d]); ss += x[d] * x[d]; }
    float inv = rsqrtf(ss * (1.0f / 64.0f) + 1e-6f);
    #pragma unroll
    for (int d = 0; d < 64; ++d) x[d] *= inv * mkw[d];
    rope_apply(x, (float)tk);
    #pragma unroll
    for (int d = 0; d < 64; ++d) kk[(h * 16 + tk) * 64 + d] = f2bf(x[d]);
    #pragma unroll
    for (int d = 0; d < 64; ++d) vv[(h * 16 + tk) * 64 + d] = qkv[rowb + 2048 + h * 64 + d];
  }
  __syncthreads();
  int tq = tk;
  float q[64]; float ss = 0.f;
  #pragma unroll
  for (int d = 0; d < 64; ++d) { q[d] = bf2f(qkv[rowb + h * 64 + d]); ss += q[d] * q[d]; }
  float inv = rsqrtf(ss * (1.0f / 64.0f) + 1e-6f);
  #pragma unroll
  for (int d = 0; d < 64; ++d) q[d] *= inv * mqw[d];
  rope_apply(q, (float)tq);

  float sc[16]; float mx = -1e30f;
  #pragma unroll
  for (int j = 0; j < 16; ++j) {
    float dot = 0.f;
    #pragma unroll
    for (int d = 0; d < 64; ++d) dot += q[d] * bf2f(kk[(h * 16 + j) * 64 + d]);
    int dt = tq - j; dt = dt < 0 ? -dt : dt;
    sc[j] = (dt <= LOCAL_W) ? dot * 0.125f : -1e9f;
    mx = fmaxf(mx, sc[j]);
  }
  float sum = 0.f;
  #pragma unroll
  for (int j = 0; j < 16; ++j) { sc[j] = __expf(sc[j] - mx); sum += sc[j]; }
  float isum = 1.0f / sum;
  float o[64] = {0.f};
  #pragma unroll
  for (int j = 0; j < 16; ++j) {
    float p = sc[j] * isum;
    #pragma unroll
    for (int d = 0; d < 64; ++d) o[d] += p * bf2f(vv[(h * 16 + j) * 64 + d]);
  }
  size_t ob = (size_t)(s * 16 + tq) * CM_C + h * 64;
  #pragma unroll
  for (int d = 0; d < 64; ++d) out[ob + d] = f2bf(o[d]);
}

// ---------------- attention 2: thread = (L-row, head), 16 keys ----------------
__global__ __launch_bounds__(256) void attn2_kernel(const unsigned short* __restrict__ q2,
                                                    const float* __restrict__ k2r, const float* __restrict__ v2,
                                                    const float* __restrict__ kqw,
                                                    unsigned short* __restrict__ out) {
  int gid = blockIdx.x * 256 + threadIdx.x;   // L*16
  int row = gid >> 4, h = gid & 15;
  int t = row / S_DIM;                        // query frame
  size_t qb = (size_t)row * CM_C + h * 64;
  float q[64]; float ss = 0.f;
  #pragma unroll
  for (int d = 0; d < 64; ++d) { q[d] = bf2f(q2[qb + d]); ss += q[d] * q[d]; }
  float inv = rsqrtf(ss * (1.0f / 64.0f) + 1e-6f);
  #pragma unroll
  for (int d = 0; d < 64; ++d) q[d] *= inv * kqw[d];
  rope_apply(q, (float)t);

  float sc[16]; float mx = -1e30f;
  #pragma unroll
  for (int j = 0; j < 16; ++j) {
    const float* kp = k2r + (j * 16 + h) * 64;
    float dot = 0.f;
    #pragma unroll
    for (int d = 0; d < 64; ++d) dot += q[d] * kp[d];
    int dt = t - j; dt = dt < 0 ? -dt : dt;
    sc[j] = (dt <= LOCAL_W) ? dot * 0.125f : -1e9f;
    mx = fmaxf(mx, sc[j]);
  }
  float sum = 0.f;
  #pragma unroll
  for (int j = 0; j < 16; ++j) { sc[j] = __expf(sc[j] - mx); sum += sc[j]; }
  float isum = 1.0f / sum;
  float o[64] = {0.f};
  #pragma unroll
  for (int j = 0; j < 16; ++j) {
    float p = sc[j] * isum;
    const float* vp = v2 + (j * 16 + h) * 64;
    #pragma unroll
    for (int d = 0; d < 64; ++d) o[d] += p * vp[d];
  }
  #pragma unroll
  for (int d = 0; d < 64; ++d) out[qb + d] = f2bf(o[d]);
}

// ---------------- host launcher ----------------
extern "C" void kernel_launch(void* const* d_in, const int* in_sizes, int n_in,
                              void* d_out, int out_size, void* d_ws, size_t ws_size,
                              hipStream_t stream) {
  (void)in_sizes; (void)n_in; (void)out_size; (void)ws_size;
  const float* x      = (const float*)d_in[0];
  const float* mouse  = (const float*)d_in[1];
  const float* kbc    = (const float*)d_in[2];
  const float* kb_w1  = (const float*)d_in[3];
  const float* kb_b1  = (const float*)d_in[4];
  const float* kb_w2  = (const float*)d_in[5];
  const float* kb_b2  = (const float*)d_in[6];
  const float* mm_w1  = (const float*)d_in[7];
  const float* mm_b1  = (const float*)d_in[8];
  const float* mm_w2  = (const float*)d_in[9];
  const float* mm_b2  = (const float*)d_in[10];
  const float* ln_g   = (const float*)d_in[11];
  const float* ln_b   = (const float*)d_in[12];
  const float* tqkv_w = (const float*)d_in[13];
  const float* mq_w   = (const float*)d_in[14];
  const float* mk_w   = (const float*)d_in[15];
  const float* pm_w   = (const float*)d_in[16];
  const float* kq_w   = (const float*)d_in[17];
  const float* kkv_w  = (const float*)d_in[18];
  const float* kqn_w  = (const float*)d_in[19];
  const float* kkn_w  = (const float*)d_in[20];
  const float* pk_w   = (const float*)d_in[21];
  float* out = (float*)d_out;

  size_t off = 0;
  char* wsb = (char*)d_ws;
  auto take = [&](size_t bytes) -> char* {
    char* p = wsb + off; off += (bytes + 255) & ~(size_t)255; return p;
  };
  // transposed bf16 weights: Wt[N][K]
  unsigned short* W1t   = (unsigned short*)take((size_t)1024 * 1536 * 2);  // K=1536 (rows 0..1535 of mm_w1)
  unsigned short* W2t   = (unsigned short*)take((size_t)1024 * 1024 * 2);
  unsigned short* Wqkvt = (unsigned short*)take((size_t)3072 * 1024 * 2);
  unsigned short* Wpmt  = (unsigned short*)take((size_t)1536 * 1024 * 2);
  unsigned short* Wkqt  = (unsigned short*)take((size_t)1024 * 1536 * 2);
  unsigned short* Wpkt  = (unsigned short*)take((size_t)1536 * 1024 * 2);
  float* tbias = (float*)take((size_t)16 * 1024 * 4);
  float* kbh   = (float*)take((size_t)61 * 128 * 4);
  float* kbv   = (float*)take((size_t)61 * 128 * 4);
  float* k2raw = (float*)take((size_t)16 * 16 * 64 * 4);
  float* k2r   = (float*)take((size_t)16 * 16 * 64 * 4);
  float* v2    = (float*)take((size_t)16 * 16 * 64 * 4);
  unsigned short* xb   = (unsigned short*)take((size_t)L_DIM * IMG_C * 2);  // reused as x2b
  unsigned short* h1b  = (unsigned short*)take((size_t)L_DIM * CM_C * 2);   // reused as attoutb
  unsigned short* h2b  = (unsigned short*)take((size_t)L_DIM * CM_C * 2);   // reused as out2b
  unsigned short* hb   = (unsigned short*)take((size_t)L_DIM * CM_C * 2);   // reused as q2b
  unsigned short* qkvb = (unsigned short*)take((size_t)L_DIM * QKV_C * 2);
  unsigned short* x2b = xb, *attoutb = h1b, *out2b = h2b, *q2b = hb;

  auto cvtT = [&](const float* s, unsigned short* d, int K, int N) {
    int n = K * N;
    cvt_transpose_kernel<<<(n + 255) / 256, 256, 0, stream>>>(s, d, K, N);
  };
  // NOTE: mm_w1 has 1560 rows; transpose reads only rows 0..1535 (K=1536) —
  // rows 1536..1559 are folded into tbias.
  cvtT(mm_w1,  W1t,   1536, 1024);
  cvtT(mm_w2,  W2t,   1024, 1024);
  cvtT(tqkv_w, Wqkvt, 1024, 3072);
  cvtT(pm_w,   Wpmt,  1024, 1536);
  cvtT(kq_w,   Wkqt,  1536, 1024);
  cvtT(pk_w,   Wpkt,  1024, 1536);

  permute_x_kernel<<<(L_DIM * IMG_C + 255) / 256, 256, 0, stream>>>(x, xb);
  tbias_kernel<<<64, 256, 0, stream>>>(mouse, mm_w1, mm_b1, tbias);
  kb1_kernel<<<31, 256, 0, stream>>>(kbc, kb_w1, kb_b1, kbh);
  kb2_kernel<<<31, 256, 0, stream>>>(kbh, kb_w2, kb_b2, kbv);
  kv_kernel<<<128, 256, 0, stream>>>(kbv, kkv_w, k2raw, v2);
  k2rope_kernel<<<1, 256, 0, stream>>>(k2raw, kkn_w, k2r);

  const int MB = L_DIM / 128;   // 195
  // G1: h1 = gelu(xb @ W1[:1536] + tbias[t])
  gemm_bf16_wmma<1, 0, false, true><<<dim3(CM_C / 128, MB), 256, 0, stream>>>(
      xb, W1t, nullptr, tbias, h1b, nullptr, nullptr, L_DIM, CM_C, IMG_C);
  // G2: h2 = h1 @ mm_w2 + mm_b2
  gemm_bf16_wmma<0, 0, true, false><<<dim3(CM_C / 128, MB), 256, 0, stream>>>(
      h1b, W2t, mm_b2, nullptr, h2b, nullptr, nullptr, L_DIM, CM_C, CM_C);
  // LayerNorm
  ln_kernel<<<L_DIM, 256, 0, stream>>>(h2b, ln_g, ln_b, hb);
  // G3: qkv = h @ tqkv_w
  gemm_bf16_wmma<0, 0, false, false><<<dim3(QKV_C / 128, MB), 256, 0, stream>>>(
      hb, Wqkvt, nullptr, nullptr, qkvb, nullptr, nullptr, L_DIM, QKV_C, CM_C);
  // attention 1
  attn1_kernel<<<S_DIM, 256, 0, stream>>>(qkvb, mq_w, mk_w, attoutb);
  // G4: x2 = x + attout @ proj_mouse_w   (row remap (s,t)->t*S+s), f32 -> d_out, bf16 -> x2b
  gemm_bf16_wmma<0, 1, false, false><<<dim3(IMG_C / 128, MB), 256, 0, stream>>>(
      attoutb, Wpmt, nullptr, nullptr, x2b, out, x, L_DIM, IMG_C, CM_C);
  // G5: q2 = x2 @ key_q_w
  gemm_bf16_wmma<0, 0, false, false><<<dim3(CM_C / 128, MB), 256, 0, stream>>>(
      x2b, Wkqt, nullptr, nullptr, q2b, nullptr, nullptr, L_DIM, CM_C, IMG_C);
  // attention 2
  attn2_kernel<<<(L_DIM * 16) / 256, 256, 0, stream>>>(q2b, k2r, v2, kqn_w, out2b);
  // G6: out = x2 + out2 @ proj_keyboard_w   (resid = d_out itself)
  gemm_bf16_wmma<0, 2, false, false><<<dim3(IMG_C / 128, MB), 256, 0, stream>>>(
      out2b, Wpkt, nullptr, nullptr, nullptr, out, out, L_DIM, IMG_C, CM_C);
}
